// Pointnet_SetAbstraction_msg_37933151158470
// MI455X (gfx1250) — matrix operand order repaired and checked by reading, not compiled, not run on hardware
//
#include <hip/hip_runtime.h>
#include <hip/hip_bf16.h>

typedef __attribute__((ext_vector_type(16))) _Float16 v16h;
typedef __attribute__((ext_vector_type(8)))  _Float16 v8h;
typedef __attribute__((ext_vector_type(8)))  float    v8f;

#define BN    8
#define NPTS  8192
#define NSAMP 1024
#define CTOT  320
#define MT    4      // M-tiles per wave in the GEMM

// ---------------------------------------------------------------------------
// Farthest point sampling: one workgroup per batch, 256 threads, 32 pts/thread.
// ---------------------------------------------------------------------------
__global__ void fps_kernel(const float* __restrict__ points,
                           float* __restrict__ centroids) {
  const int b = blockIdx.x;
  const float* P = points + (size_t)b * NPTS * 3;
  __shared__ float s_val[256];
  __shared__ int   s_idx[256];
  __shared__ float s_c[3];
  const int t = threadIdx.x;
  const int PT = NPTS / 256; // 32
  float mind[PT];
#pragma unroll
  for (int i = 0; i < PT; ++i) mind[i] = 1e10f;
  int far = 0;
  for (int i = 0; i < NSAMP; ++i) {
    if (t == 0) {
      float cx = P[far * 3 + 0], cy = P[far * 3 + 1], cz = P[far * 3 + 2];
      s_c[0] = cx; s_c[1] = cy; s_c[2] = cz;
      float* C = centroids + ((size_t)b * NSAMP + i) * 3;
      C[0] = cx; C[1] = cy; C[2] = cz;
    }
    __syncthreads();
    const float cx = s_c[0], cy = s_c[1], cz = s_c[2];
    float best = -1.0f; int bi = 0;
#pragma unroll
    for (int j = 0; j < PT; ++j) {
      const int p = t + j * 256;
      const float dx = P[p * 3 + 0] - cx;
      const float dy = P[p * 3 + 1] - cy;
      const float dz = P[p * 3 + 2] - cz;
      float d = dx * dx + dy * dy + dz * dz;
      float md = mind[j];
      md = d < md ? d : md;
      mind[j] = md;
      if (md > best) { best = md; bi = p; }
    }
    s_val[t] = best; s_idx[t] = bi;
    __syncthreads();
    for (int s2 = 128; s2 > 0; s2 >>= 1) {
      if (t < s2 && s_val[t + s2] > s_val[t]) {
        s_val[t] = s_val[t + s2]; s_idx[t] = s_idx[t + s2];
      }
      __syncthreads();
    }
    far = s_idx[0];
    __syncthreads();
  }
}

// ---------------------------------------------------------------------------
// KNN top-64 per (batch, query): d^2 cached in LDS, 64 argmin extractions.
// Ascending order => top-16/32 are prefixes of top-64.
// ---------------------------------------------------------------------------
__global__ void knn_kernel(const float* __restrict__ points,
                           const float* __restrict__ centroids,
                           int* __restrict__ idx64) {
  const int b = blockIdx.y;
  const int q = blockIdx.x;
  __shared__ float s_d[NPTS];
  __shared__ float r_val[256];
  __shared__ int   r_idx[256];
  const float* P = points + (size_t)b * NPTS * 3;
  const float* C = centroids + ((size_t)b * NSAMP + q) * 3;
  const float qx = C[0], qy = C[1], qz = C[2];
  for (int p = threadIdx.x; p < NPTS; p += 256) {
    const float dx = P[p * 3 + 0] - qx;
    const float dy = P[p * 3 + 1] - qy;
    const float dz = P[p * 3 + 2] - qz;
    s_d[p] = dx * dx + dy * dy + dz * dz;
  }
  __syncthreads();
  int* out = idx64 + ((size_t)b * NSAMP + q) * 64;
  for (int sel = 0; sel < 64; ++sel) {
    float best = 1e30f; int bi = 0;
    for (int j = threadIdx.x; j < NPTS; j += 256) {
      const float d = s_d[j];
      if (d < best) { best = d; bi = j; }
    }
    r_val[threadIdx.x] = best; r_idx[threadIdx.x] = bi;
    __syncthreads();
    for (int s2 = 128; s2 > 0; s2 >>= 1) {
      if (threadIdx.x < s2 && r_val[threadIdx.x + s2] < r_val[threadIdx.x]) {
        r_val[threadIdx.x] = r_val[threadIdx.x + s2];
        r_idx[threadIdx.x] = r_idx[threadIdx.x + s2];
      }
      __syncthreads();
    }
    if (threadIdx.x == 0) {
      const int w = r_idx[0];
      out[sel] = w;
      s_d[w] = 1e30f;
    }
    __syncthreads();
  }
}

// ---------------------------------------------------------------------------
// Gather grouped points -> fp16, K padded 3 -> 32 with zeros (64B rows) so the
// layer-0 GEMM needs no bounds checks and gets aligned b128 loads.
// ---------------------------------------------------------------------------
__global__ void gather_kernel(const float* __restrict__ points,
                              const int* __restrict__ idx64,
                              _Float16* __restrict__ X, int k) {
  const size_t total = (size_t)BN * NSAMP * k;
  for (size_t r = (size_t)blockIdx.x * blockDim.x + threadIdx.x; r < total;
       r += (size_t)gridDim.x * blockDim.x) {
    const int    j  = (int)(r % k);
    const size_t sq = r / k;
    const int    s  = (int)(sq % NSAMP);
    const int    b  = (int)(sq / NSAMP);
    const int pid = idx64[((size_t)b * NSAMP + s) * 64 + j];
    const float* p = points + ((size_t)b * NPTS + pid) * 3;
    v8h v0 = {};
    v0[0] = (_Float16)p[0]; v0[1] = (_Float16)p[1]; v0[2] = (_Float16)p[2];
    v8h z = {};
    v8h* x = (v8h*)(X + r * 32);
    x[0] = v0; x[1] = z; x[2] = z; x[3] = z;
  }
}

// ---------------------------------------------------------------------------
// WMMA GEMM, K is a COMPILE-TIME constant (32/64/96) so the chunk loop fully
// unrolls into straight-line code: all fragment loads become independent and
// the scheduler can clause them ahead of the back-to-back wmmas.
//  - A: rows x K fp16 (pre-normalized; zero-padded for layer 0)
//  - W: N x Kw fp32; 16-column tile staged into LDS as fp16, zero-padded to K
//  - 8 waves/block, MT=4 16x16 M-tiles per wave sharing each B fragment.
// ---------------------------------------------------------------------------
template <int K>
__global__ void gemm_wmma(const _Float16* __restrict__ A,
                          const float* __restrict__ W,
                          _Float16* __restrict__ Y,
                          int Kw, int N) {
  __shared__ __align__(32) _Float16 s_w[16 * K];
  const int n0 = blockIdx.y * 16;
  for (int idx = threadIdx.x; idx < 16 * K; idx += blockDim.x) {
    const int n = idx / K, kk = idx % K;
    s_w[idx] = (kk < Kw) ? (_Float16)W[(size_t)(n0 + n) * Kw + kk]
                         : (_Float16)0.0f;
  }
  __syncthreads();
  const int lane = threadIdx.x & 31;
  const int wave = threadIdx.x >> 5;
  const int hi   = lane >> 4;            // half-wave select
  const int rlo  = lane & 15;
  const int mbase = (blockIdx.x * 8 + wave) * (16 * MT);
  const _Float16* aptr = A + (size_t)(mbase + rlo) * K + hi * 8;   // A-frag rows
  const _Float16* bptr = s_w + (size_t)rlo * K + hi * 8;           // B-frag cols
  constexpr int NC = K / 32;             // K-chunks (fully unrolled)
  v8f acc[MT] = {};
#pragma unroll
  for (int c = 0; c < NC; ++c) {
    const int k0 = c * 32;
    // ---- load phase: B fragment + all MT A fragments ----
    const v8h blo = *(const v8h*)(bptr + k0);        // K = k0 + hi*8 + 0..7
    const v8h bhi = *(const v8h*)(bptr + k0 + 16);   // K = k0 + 16 + hi*8 + 0..7
    v16h b;
#pragma unroll
    for (int e = 0; e < 8; ++e) { b[e] = blo[e]; b[8 + e] = bhi[e]; }
    v16h a[MT];
#pragma unroll
    for (int tt = 0; tt < MT; ++tt) {
      const _Float16* ap = aptr + (size_t)tt * 16 * K + k0;
      const v8h alo = *(const v8h*)(ap);        // K = k0 + hi*8 + 0..7
      const v8h ahi = *(const v8h*)(ap + 16);   // K = k0 + 16 + hi*8 + 0..7
#pragma unroll
      for (int e = 0; e < 8; ++e) { a[tt][e] = alo[e]; a[tt][8 + e] = ahi[e]; }
    }
    // ---- compute phase: MT back-to-back wmmas sharing one B fragment ----
#pragma unroll
    for (int tt = 0; tt < MT; ++tt) {
      acc[tt] = __builtin_amdgcn_wmma_f32_16x16x32_f16(
          false, a[tt], false, b, (short)0, acc[tt], false, false);
    }
  }
  const int ncol = n0 + rlo;
#pragma unroll
  for (int tt = 0; tt < MT; ++tt) {
    const int m0 = mbase + tt * 16 + hi * 8;   // lanes 0-15: M=r, 16-31: M=8+r
#pragma unroll
    for (int r = 0; r < 8; ++r)
      Y[(size_t)(m0 + r) * N + ncol] = (_Float16)acc[tt][r];
  }
}

// ---------------------------------------------------------------------------
// Batch-norm statistics: per-channel sum & sum-of-squares.
// ---------------------------------------------------------------------------
__global__ void zero_stats(float* __restrict__ sums, int n) {
  const int i = blockIdx.x * blockDim.x + threadIdx.x;
  if (i < n) sums[i] = 0.0f;
}

__global__ void colsum_kernel(const _Float16* __restrict__ Y,
                              float* __restrict__ sums, size_t rows, int C) {
  __shared__ float s_s[128];
  __shared__ float s_q[128];
  for (int c = threadIdx.x; c < C; c += blockDim.x) { s_s[c] = 0.0f; s_q[c] = 0.0f; }
  __syncthreads();
  const size_t total  = rows * (size_t)C;
  const size_t stride = (size_t)gridDim.x * blockDim.x;
  const int    smod   = (int)(stride % (size_t)C);
  size_t i = (size_t)blockIdx.x * blockDim.x + threadIdx.x;
  int    c = (int)(i % (size_t)C);
  for (; i < total; i += stride) {
    const float v = (float)Y[i];
    atomicAdd(&s_s[c], v);
    atomicAdd(&s_q[c], v * v);
    c += smod; if (c >= C) c -= C;
  }
  __syncthreads();
  for (int cc = threadIdx.x; cc < C; cc += blockDim.x) {
    atomicAdd(&sums[cc], s_s[cc]);
    atomicAdd(&sums[C + cc], s_q[cc]);
  }
}

__global__ void finalize_kernel(const float* __restrict__ sums,
                                const float* __restrict__ g,
                                const float* __restrict__ be,
                                float* __restrict__ scale,
                                float* __restrict__ shift,
                                float inv_count, int C) {
  const int c = threadIdx.x;
  if (c >= C) return;
  const float m = sums[c] * inv_count;
  const float v = sums[C + c] * inv_count - m * m;
  const float rs = rsqrtf(v + 1e-5f);
  const float sc = rs * g[c];
  scale[c] = sc;
  shift[c] = be[c] - m * sc;
}

// ---------------------------------------------------------------------------
// In-place BN + ReLU on an fp16 buffer (vectorized 8 halfs per iteration;
// C is a multiple of 8 so each vector stays within one row's channel run).
// ---------------------------------------------------------------------------
__global__ void normrelu_kernel(_Float16* __restrict__ Y,
                                const float* __restrict__ scale,
                                const float* __restrict__ shift,
                                size_t rows, int C) {
  __shared__ float s_sc[128];
  __shared__ float s_sh[128];
  for (int c = threadIdx.x; c < C; c += blockDim.x) {
    s_sc[c] = scale[c]; s_sh[c] = shift[c];
  }
  __syncthreads();
  const size_t total8 = rows * (size_t)C / 8;
  const size_t stride = (size_t)gridDim.x * blockDim.x;
  const int    C8     = C / 8;
  const int    smod   = (int)(stride % (size_t)C8);
  size_t i  = (size_t)blockIdx.x * blockDim.x + threadIdx.x;
  int    c8 = (int)(i % (size_t)C8);
  for (; i < total8; i += stride) {
    v8h v = ((v8h*)Y)[i];
    const int c0 = c8 * 8;
#pragma unroll
    for (int e = 0; e < 8; ++e) {
      float f = (float)v[e] * s_sc[c0 + e] + s_sh[c0 + e];
      v[e] = (_Float16)(f > 0.0f ? f : 0.0f);
    }
    ((v8h*)Y)[i] = v;
    c8 += smod; if (c8 >= C8) c8 -= C8;
  }
}

// ---------------------------------------------------------------------------
// Fused BN + ReLU + max-over-group (last layer), scatter into concat slice.
// ---------------------------------------------------------------------------
__global__ void maxpool_kernel(const _Float16* __restrict__ Y,
                               const float* __restrict__ scale,
                               const float* __restrict__ shift,
                               float* __restrict__ out,
                               int k, int C, int co) {
  const int bs = blockIdx.x;  // 0 .. B*S-1
  for (int c = threadIdx.x; c < C; c += blockDim.x) {
    const float sc = scale[c], sh = shift[c];
    float best = 0.0f;  // relu output >= 0
    for (int j = 0; j < k; ++j) {
      float v = (float)Y[((size_t)bs * k + j) * C + c] * sc + sh;
      v = v > 0.0f ? v : 0.0f;
      best = v > best ? v : best;
    }
    out[(size_t)bs * CTOT + co + c] = best;
  }
}

// ---------------------------------------------------------------------------
extern "C" void kernel_launch(void* const* d_in, const int* in_sizes, int n_in,
                              void* d_out, int out_size, void* d_ws, size_t ws_size,
                              hipStream_t stream) {
  (void)in_sizes; (void)n_in; (void)out_size; (void)ws_size;
  const float* points = (const float*)d_in[0];
  const float *w[3][3], *g[3][3], *be[3][3];
  int t = 1;
  for (int i = 0; i < 3; ++i)
    for (int j = 0; j < 3; ++j) {
      w[i][j]  = (const float*)d_in[t + 0];
      // bias d_in[t+1] intentionally unused: it cancels inside batch-norm
      g[i][j]  = (const float*)d_in[t + 2];
      be[i][j] = (const float*)d_in[t + 3];
      t += 4;
    }
  float* out = (float*)d_out;

  // Workspace layout
  auto al = [](size_t x) { return (x + 255) & ~(size_t)255; };
  char* ws = (char*)d_ws;
  size_t off = 0;
  float* centroids = (float*)(ws + off); off = al(off + (size_t)BN * NSAMP * 3 * 4);
  int*   idx64     = (int*)  (ws + off); off = al(off + (size_t)BN * NSAMP * 64 * 4);
  float* sums      = (float*)(ws + off); off = al(off + 256 * 4);
  float* scalep    = (float*)(ws + off); off = al(off + 128 * 4);
  float* shiftp    = (float*)(ws + off); off = al(off + 128 * 4);
  _Float16* bufA   = (_Float16*)(ws + off); off = al(off + (size_t)BN * NSAMP * 64 * 96 * 2);  // 96 MB
  _Float16* bufB   = (_Float16*)(ws + off);                                                    // 128 MB

  // 1) FPS + 2) KNN (top-64 once; top-16/32 are prefixes)
  fps_kernel<<<BN, 256, 0, stream>>>(points, centroids);
  knn_kernel<<<dim3(NSAMP, BN), 256, 0, stream>>>(points, centroids, idx64);

  // 3) Three MSG branches
  const int kdims[3]   = {16, 32, 64};
  const int chin[3][3] = {{32, 32, 32}, {32, 64, 64}, {32, 64, 96}};  // padded A strides
  const int chw[3][3]  = {{3, 32, 32},  {3, 64, 64},  {3, 64, 96}};   // true weight K
  const int chout[3][3]= {{32, 32, 64}, {64, 64, 128}, {64, 96, 128}};
  const int cofs[3]    = {0, 64, 192};

  for (int i = 0; i < 3; ++i) {
    const int k = kdims[i];
    const size_t rows = (size_t)BN * NSAMP * k;
    gather_kernel<<<2048, 256, 0, stream>>>(points, idx64, bufA, k);

    const _Float16* src[3] = {bufA, bufB, bufA};
    _Float16*       dst[3] = {bufB, bufA, bufB};
    for (int j = 0; j < 3; ++j) {
      const int K = chin[i][j], Kw = chw[i][j], Cout = chout[i][j];
      dim3 grid((unsigned)(rows / (16 * MT * 8)), (unsigned)(Cout / 16));
      if (K == 32) {
        gemm_wmma<32><<<grid, 256, 0, stream>>>(src[j], w[i][j], dst[j], Kw, Cout);
      } else if (K == 64) {
        gemm_wmma<64><<<grid, 256, 0, stream>>>(src[j], w[i][j], dst[j], Kw, Cout);
      } else {
        gemm_wmma<96><<<grid, 256, 0, stream>>>(src[j], w[i][j], dst[j], Kw, Cout);
      }
      zero_stats<<<1, 256, 0, stream>>>(sums, 256);
      colsum_kernel<<<1024, 256, 0, stream>>>(dst[j], sums, rows, Cout);
      finalize_kernel<<<1, 128, 0, stream>>>(sums, g[i][j], be[i][j],
                                             scalep, shiftp,
                                             1.0f / (float)rows, Cout);
      if (j < 2) {
        normrelu_kernel<<<1024, 256, 0, stream>>>(dst[j], scalep, shiftp, rows, Cout);
      }
    }
    maxpool_kernel<<<BN * NSAMP, 128, 0, stream>>>(bufB, scalep, shiftp, out,
                                                   k, chout[i][2], cofs[i]);
  }
}
